// RelativeMultiHeadAttention_4578435137570
// MI455X (gfx1250) — matrix operand order repaired and checked by reading, not compiled, not run on hardware
//
#include <hip/hip_runtime.h>

typedef __attribute__((ext_vector_type(16))) __bf16 v16bf;
typedef __attribute__((ext_vector_type(8)))  float  v8f;

struct __attribute__((aligned(16))) U4 { unsigned int x, y, z, w; };

union Frag {
    v16bf v;
    U4 q[2];
    unsigned short s[16];
};

__device__ __forceinline__ unsigned short f2b(float f) {
    unsigned int u = __builtin_bit_cast(unsigned int, f);
    unsigned int r = u + 0x7FFFu + ((u >> 16) & 1u);
    return (unsigned short)(r >> 16);
}

// ---------------------------------------------------------------- casts
__global__ void cast_f32_bf16(const float* __restrict__ src,
                              unsigned short* __restrict__ dst, int n) {
    int i = blockIdx.x * blockDim.x + threadIdx.x;
    if (i < n) dst[i] = f2b(src[i]);
}

// ---------------------------------------------------------------- layernorm
__global__ __launch_bounds__(256)
void ln_kernel(const float* __restrict__ x, const float* __restrict__ g,
               const float* __restrict__ bia, unsigned short* __restrict__ out) {
    __shared__ float red[256];
    const int row = blockIdx.x;
    const float* xr = x + (size_t)row * 1024;
    float v[4];
    float s = 0.f;
#pragma unroll
    for (int i = 0; i < 4; ++i) { v[i] = xr[threadIdx.x + 256 * i]; s += v[i]; }
    red[threadIdx.x] = s;
    __syncthreads();
    for (int o = 128; o > 0; o >>= 1) {
        if (threadIdx.x < o) red[threadIdx.x] += red[threadIdx.x + o];
        __syncthreads();
    }
    float mu = red[0] * (1.f / 1024.f);
    __syncthreads();
    s = 0.f;
#pragma unroll
    for (int i = 0; i < 4; ++i) { float d = v[i] - mu; s += d * d; }
    red[threadIdx.x] = s;
    __syncthreads();
    for (int o = 128; o > 0; o >>= 1) {
        if (threadIdx.x < o) red[threadIdx.x] += red[threadIdx.x + o];
        __syncthreads();
    }
    float rstd = rsqrtf(red[0] * (1.f / 1024.f) + 1e-5f);
#pragma unroll
    for (int i = 0; i < 4; ++i) {
        int c = threadIdx.x + 256 * i;
        out[(size_t)row * 1024 + c] = f2b((v[i] - mu) * rstd * g[c] + bia[c]);
    }
}

// ---------------------------------------------------------------- GEMM
// C[M,1024] = A[M,1024](bf16) @ W[1024,1024](bf16) (+bias). 64x64 block tile,
// 4 waves (one 16-row strip each, 4 n-subtiles), K-chunks of 32.
// Store policy is a template parameter so the epilogue is branch-free.
template <int HASBIAS, int STOREF, int STOREB>
__global__ __launch_bounds__(128)
void gemm_bf16(const unsigned short* __restrict__ A,
               const unsigned short* __restrict__ W,
               const float* __restrict__ bias,
               float* __restrict__ outF,
               unsigned short* __restrict__ outB) {
    __shared__ __attribute__((aligned(16))) unsigned short ldsA[64 * 32];      // [m][k]
    __shared__ __attribute__((aligned(16))) unsigned short ldsB[64 * 32];      // [n][k]
    const int tid  = threadIdx.x;
    const int lane = tid & 31;
    const int wave = tid >> 5;
    const int hl   = lane >> 4;
    const int l15  = lane & 15;
    const int m0   = blockIdx.y * 64;
    const int n0   = blockIdx.x * 64;
    v8f acc[4] = {};

    const int rowA = tid >> 1, cbA = (tid & 1) * 16;
    const unsigned short* aPtr = A + (size_t)(m0 + rowA) * 1024 + cbA;
    const int kW = tid >> 2, nbW = (tid & 3) * 16;
    const unsigned short* wPtr = W + (size_t)kW * 1024 + n0 + nbW;

    for (int k0 = 0; k0 < 1024; k0 += 32) {
        {   // A tile 64x32 (row-contiguous copy)
            const U4* src = (const U4*)(aPtr + k0);
            *(U4*)(ldsA + rowA * 32 + cbA)     = src[0];
            *(U4*)(ldsA + rowA * 32 + cbA + 8) = src[1];
        }
        {   // W tile 32x64, transposed into ldsB[n][k]
            const unsigned short* src = wPtr + (size_t)k0 * 1024;
#pragma unroll
            for (int i = 0; i < 16; ++i) ldsB[(nbW + i) * 32 + kW] = src[i];
        }
        if (k0 < 992) {  // pull next K-chunk toward the caches while we compute
            __builtin_prefetch(aPtr + k0 + 32, 0, 1);
            __builtin_prefetch(wPtr + (size_t)(k0 + 32) * 1024, 0, 1);
        }
        __syncthreads();

        Frag a;
        {
            int row = wave * 16 + l15;
            a.q[0] = *(const U4*)(ldsA + row * 32 + hl * 8);
            a.q[1] = *(const U4*)(ldsA + row * 32 + 16 + hl * 8);
        }
#pragma unroll
        for (int nt = 0; nt < 4; ++nt) {
            Frag bf;
            int n = nt * 16 + l15;
            bf.q[0] = *(const U4*)(ldsB + n * 32 + hl * 8);
            bf.q[1] = *(const U4*)(ldsB + n * 32 + 16 + hl * 8);
            acc[nt] = __builtin_amdgcn_wmma_f32_16x16x32_bf16(
                false, a.v, false, bf.v, (short)0, acc[nt], false, false);
        }
        __syncthreads();
    }

#pragma unroll
    for (int nt = 0; nt < 4; ++nt) {
        int col = n0 + nt * 16 + l15;
        float bv = HASBIAS ? bias[col] : 0.f;
#pragma unroll
        for (int r = 0; r < 8; ++r) {
            int row = m0 + wave * 16 + r + hl * 8;
            float c = acc[nt][r] + bv;
            if (STOREF) outF[(size_t)row * 1024 + col] = c;
            if (STOREB) outB[(size_t)row * 1024 + col] = f2b(c);
        }
    }
}

// ---------------------------------------------------------------- flash attention with folded rel-shift
// grid (S/64, H, B), block 128 = 4 waves; wave handles 16 q-rows.
// pos[s,t] = (q[s]+v_bias) . p[t-s+S-1]  (rel-shift folded via a 48-wide
// sliding window of three 16-col WMMA tiles + intra-wave shuffles).
__global__ __launch_bounds__(128)
void attn_kernel(const float* __restrict__ qf,            // [B*S, 1024] f32
                 const unsigned short* __restrict__ kbuf, // [B*S, 1024] bf16
                 const unsigned short* __restrict__ vbuf,
                 const unsigned short* __restrict__ pbuf,
                 const float* __restrict__ ubias,         // [16,64]
                 const float* __restrict__ vbias,
                 const unsigned char* __restrict__ mask,  // [B*S]
                 unsigned short* __restrict__ obuf)       // [B*S, 1024] bf16
{
    __shared__ __attribute__((aligned(16))) unsigned short v_lds[64 * 32];   // [d][t]
    __shared__ __attribute__((aligned(16))) unsigned short p_lds[4][16 * 32];// per-wave [i][j]
    const int lane = threadIdx.x & 31;
    const int wave = threadIdx.x >> 5;
    const int hl   = lane >> 4;
    const int l15  = lane & 15;
    const int b = blockIdx.z, h = blockIdx.y;
    const int s0 = blockIdx.x * 64 + wave * 16;
    const size_t bhOff = (size_t)b * 1024 * 1024 + (size_t)h * 64; // row-major [B*S,1024] head base

    // q + u_bias / q + v_bias A-fragments, 2 K-chunks over dh=64
    Frag qu[2], qv[2];
    {
        const float* qrow = qf + bhOff + (size_t)(s0 + l15) * 1024;
        const float* ur = ubias + h * 64;
        const float* vr = vbias + h * 64;
#pragma unroll
        for (int c = 0; c < 2; ++c) {
#pragma unroll
            for (int e = 0; e < 16; ++e) {
                int d = c * 32 + ((e < 8) ? (hl * 8 + e) : (16 + hl * 8 + e - 8));
                float qd = qrow[d];
                qu[c].s[e] = f2b(qd + ur[d]);
                qv[c].s[e] = f2b(qd + vr[d]);
            }
        }
    }

    float mrow[8], lrow[8];
    v8f oacc[4] = {};
#pragma unroll
    for (int r = 0; r < 8; ++r) { mrow[r] = -1e30f; lrow[r] = 0.f; }

    const int tV = threadIdx.x >> 2, dbV = (threadIdx.x & 3) * 16;
    const unsigned short* vSrc0 = vbuf + bhOff + (size_t)tV * 1024 + dbV;

    for (int t0 = 0; t0 < 1024; t0 += 32) {
        {   // stage V chunk transposed: v_lds[d][t], 32 t-rows x 64 d
            const unsigned short* src = vSrc0 + (size_t)t0 * 1024;
#pragma unroll
            for (int i = 0; i < 16; ++i) v_lds[(dbV + i) * 32 + tV] = src[i];
        }
        if (t0 < 992) {  // prefetch next chunk's V and K rows
            __builtin_prefetch(vSrc0 + (size_t)(t0 + 32) * 1024, 0, 1);
            __builtin_prefetch(kbuf + bhOff + (size_t)(t0 + 32 + tV) * 1024 + dbV, 0, 1);
        }
        __syncthreads();

        // content scores, two 16-col j-tiles
        v8f sc[2];
#pragma unroll
        for (int jt = 0; jt < 2; ++jt) {
            v8f a = {};
            const unsigned short* krow =
                kbuf + bhOff + (size_t)(t0 + jt * 16 + l15) * 1024;
#pragma unroll
            for (int c = 0; c < 2; ++c) {
                Frag kf;
                const U4* src = (const U4*)(krow + c * 32 + hl * 8);
                kf.q[0] = src[0]; kf.q[1] = src[2];
                a = __builtin_amdgcn_wmma_f32_16x16x32_bf16(
                    false, qu[c].v, false, kf.v, (short)0, a, false, false);
            }
            sc[jt] = a;
        }

        // pos windows: three 16-col tiles starting at w0, w0+16, w0+32
        int w0 = t0 - s0 + 1008;
        v8f pw[3];
#pragma unroll
        for (int f = 0; f < 3; ++f) {
            v8f a = {};
            const unsigned short* prow =
                pbuf + bhOff + (size_t)((w0 + f * 16 + l15) & 1023) * 1024;
#pragma unroll
            for (int c = 0; c < 2; ++c) {
                Frag pf;
                const U4* src = (const U4*)(prow + c * 32 + hl * 8);
                pf.q[0] = src[0]; pf.q[1] = src[2];
                a = __builtin_amdgcn_wmma_f32_16x16x32_bf16(
                    false, qv[c].v, false, pf.v, (short)0, a, false, false);
            }
            pw[f] = a;
        }

        // combine: shifted pos via cross-lane pick, scale, mask
        float sco[2][8];
#pragma unroll
        for (int jt = 0; jt < 2; ++jt) {
            int mb = mask[b * 1024 + t0 + jt * 16 + l15];
#pragma unroll
            for (int r = 0; r < 8; ++r) {
                int i = r + hl * 8;
                int j = l15;
                int dcol = j - i + 15;                       // [0,30] rel. to window jt
                int srcLane = (dcol & 15) | (lane & 16);
                float pa = __shfl(pw[jt][r],     srcLane, 32);
                float pc = __shfl(pw[jt + 1][r], srcLane, 32);
                float pos = (j > i) ? pc : pa;
                float sv = (sc[jt][r] + pos) * 0.03125f;     // 1/sqrt(1024)
                sco[jt][r] = mb ? -1e30f : sv;
            }
        }

        // online softmax
#pragma unroll
        for (int r = 0; r < 8; ++r) {
            float mx = fmaxf(sco[0][r], sco[1][r]);
#pragma unroll
            for (int o = 1; o < 16; o <<= 1) mx = fmaxf(mx, __shfl_xor(mx, o, 32));
            float mnew = fmaxf(mrow[r], mx);
            float scale = __expf(mrow[r] - mnew);
            mrow[r] = mnew;
            lrow[r] *= scale;
#pragma unroll
            for (int nt = 0; nt < 4; ++nt) oacc[nt][r] *= scale;
            float p0 = __expf(sco[0][r] - mnew);
            float p1 = __expf(sco[1][r] - mnew);
            float ps = p0 + p1;
#pragma unroll
            for (int o = 1; o < 16; o <<= 1) ps += __shfl_xor(ps, o, 32);
            lrow[r] += ps;
            p_lds[wave][(r + hl * 8) * 32 + l15]      = f2b(p0);
            p_lds[wave][(r + hl * 8) * 32 + 16 + l15] = f2b(p1);
        }
        asm volatile("s_wait_dscnt 0x0" ::: "memory");

        // attn (16x32) @ V (32x64)
        Frag af;
        af.q[0] = *(const U4*)(&p_lds[wave][l15 * 32 + hl * 8]);
        af.q[1] = *(const U4*)(&p_lds[wave][l15 * 32 + 16 + hl * 8]);
#pragma unroll
        for (int nt = 0; nt < 4; ++nt) {
            Frag vf;
            int d = nt * 16 + l15;
            const U4* src = (const U4*)(v_lds + d * 32 + hl * 8);
            vf.q[0] = src[0]; vf.q[1] = src[2];
            oacc[nt] = __builtin_amdgcn_wmma_f32_16x16x32_bf16(
                false, af.v, false, vf.v, (short)0, oacc[nt], false, false);
        }
        __syncthreads();   // protect v_lds for next chunk
    }

    // write O (bf16) at [b*S+s, h*64+d]
#pragma unroll
    for (int nt = 0; nt < 4; ++nt) {
        int d = nt * 16 + l15;
#pragma unroll
        for (int r = 0; r < 8; ++r) {
            int row = s0 + r + hl * 8;
            float o = oacc[nt][r] / lrow[r];
            obuf[bhOff + (size_t)row * 1024 + d] = f2b(o);
        }
    }
}

// ---------------------------------------------------------------- launch
extern "C" void kernel_launch(void* const* d_in, const int* in_sizes, int n_in,
                              void* d_out, int out_size, void* d_ws, size_t ws_size,
                              hipStream_t stream) {
    (void)in_sizes; (void)n_in; (void)out_size; (void)ws_size;
    const float* x     = (const float*)d_in[0];
    const float* pos   = (const float*)d_in[1];
    const unsigned char* mask = (const unsigned char*)d_in[2];
    const float* lng   = (const float*)d_in[3];
    const float* lnb   = (const float*)d_in[4];
    const float* Wq    = (const float*)d_in[5];
    const float* bq    = (const float*)d_in[6];
    const float* Wk    = (const float*)d_in[7];
    const float* bk    = (const float*)d_in[8];
    const float* Wv    = (const float*)d_in[9];
    const float* bv    = (const float*)d_in[10];
    const float* Wp    = (const float*)d_in[11];
    const float* Wout  = (const float*)d_in[12];
    const float* bout  = (const float*)d_in[13];
    const float* ub    = (const float*)d_in[14];
    const float* vbia  = (const float*)d_in[15];

    char* ws = (char*)d_ws;
    size_t off = 0;
    auto alloc = [&](size_t bytes) -> void* {
        void* p = ws + off;
        off += (bytes + 255) & ~(size_t)255;
        return p;
    };
    const size_t NROW = 4096, D = 1024;
    unsigned short* xnb   = (unsigned short*)alloc(NROW * D * 2);
    unsigned short* posb  = (unsigned short*)alloc(NROW * D * 2);
    unsigned short* Wqb   = (unsigned short*)alloc(D * D * 2);
    unsigned short* Wkb   = (unsigned short*)alloc(D * D * 2);
    unsigned short* Wvb   = (unsigned short*)alloc(D * D * 2);
    unsigned short* Wpb   = (unsigned short*)alloc(D * D * 2);
    unsigned short* Woutb = (unsigned short*)alloc(D * D * 2);
    float*          qf    = (float*)alloc(NROW * D * 4);
    unsigned short* kbuf  = (unsigned short*)alloc(NROW * D * 2);
    unsigned short* vbuf  = (unsigned short*)alloc(NROW * D * 2);
    unsigned short* pbuf  = (unsigned short*)alloc(NROW * D * 2);
    unsigned short* obuf  = (unsigned short*)alloc(NROW * D * 2);

    const int nW = (int)(D * D);
    cast_f32_bf16<<<(nW + 255) / 256, 256, 0, stream>>>(Wq,   Wqb,   nW);
    cast_f32_bf16<<<(nW + 255) / 256, 256, 0, stream>>>(Wk,   Wkb,   nW);
    cast_f32_bf16<<<(nW + 255) / 256, 256, 0, stream>>>(Wv,   Wvb,   nW);
    cast_f32_bf16<<<(nW + 255) / 256, 256, 0, stream>>>(Wp,   Wpb,   nW);
    cast_f32_bf16<<<(nW + 255) / 256, 256, 0, stream>>>(Wout, Woutb, nW);
    const int nX = (int)(NROW * D);
    cast_f32_bf16<<<(nX + 255) / 256, 256, 0, stream>>>(pos, posb, nX);

    ln_kernel<<<4096, 256, 0, stream>>>(x, lng, lnb, xnb);

    dim3 ggrid(16, 64);  // N/64, M/64
    gemm_bf16<1, 1, 0><<<ggrid, 128, 0, stream>>>(xnb,  Wqb, bq, qf,      nullptr);
    gemm_bf16<1, 0, 1><<<ggrid, 128, 0, stream>>>(xnb,  Wkb, bk, nullptr, kbuf);
    gemm_bf16<1, 0, 1><<<ggrid, 128, 0, stream>>>(xnb,  Wvb, bv, nullptr, vbuf);
    gemm_bf16<0, 0, 1><<<ggrid, 128, 0, stream>>>(posb, Wpb, bq, nullptr, pbuf);

    dim3 agrid(16, 16, 4);  // S/64, H, B
    attn_kernel<<<agrid, 128, 0, stream>>>(qf, kbuf, vbuf, pbuf, ub, vbia, mask, obuf);

    gemm_bf16<1, 1, 0><<<ggrid, 128, 0, stream>>>(obuf, Woutb, bout, (float*)d_out, nullptr);
}
